// MemoryBank_79826262164167
// MI455X (gfx1250) — matrix operand order repaired and checked by reading, not compiled, not run on hardware
//
#include <hip/hip_runtime.h>

#define MEMN  32768
#define FDIM  128
#define BATCH 4096
#define SPLITS 8

typedef __attribute__((ext_vector_type(16))) __bf16 v16bf;
typedef __attribute__((ext_vector_type(8)))  float  v8f;
typedef __attribute__((ext_vector_type(4)))  unsigned u32x4;
typedef __attribute__((ext_vector_type(8)))  int      i32x8;
typedef __attribute__((ext_vector_type(4)))  int      i32x4;

union BfVec {
  v16bf v;
  unsigned short s[16];
  uint4 q[2];
};

__device__ __forceinline__ unsigned short f2bfu(float f) {
  union { float f; unsigned u; } c; c.f = f;
  unsigned r = (c.u + 0x7FFFu + ((c.u >> 16) & 1u)) >> 16;   // round-to-nearest-even
  return (unsigned short)r;
}

// 16 contiguous bf16 (32B, 16B-aligned)
__device__ __forceinline__ v16bf load_v16(const unsigned short* p) {
  BfVec t;
  t.q[0] = ((const uint4*)p)[0];
  t.q[1] = ((const uint4*)p)[1];
  return t.v;
}
// A-operand: two separate 16B chunks (elements 0..7 / 8..15)
__device__ __forceinline__ v16bf load_a16(const unsigned short* p0,
                                          const unsigned short* p1) {
  BfVec t;
  t.q[0] = *(const uint4*)p0;
  t.q[1] = *(const uint4*)p1;
  return t.v;
}

__device__ __forceinline__ v8f wmma_bf16(v16bf a, v16bf b, v8f c) {
  return __builtin_amdgcn_wmma_f32_16x16x32_bf16(
      false, a, false, b, (short)0, c, false, false);
}

// ---------------------------------------------------------------------------
// TDM: load a 32-row x 128-f32 tile (row stride MEMN f32) from global into
// LDS with 8-DWORD padding every 128 DWORDs (-> pitch 136 f32). 2D descriptor
// per CDNA5 ISA 8.3-8.5; issued once per wave, tracked by TENSORcnt.
// 6-arg builtin form (clang-23 / therock-10.0 toolchain).
// ---------------------------------------------------------------------------
__device__ __forceinline__ void tdm_load_tile(unsigned lds_byte_off,
                                              const void* gaddr) {
  unsigned long long ga = (unsigned long long)(uintptr_t)gaddr;
  u32x4 g0;
  g0[0] = 1u;                                            // count=1 user D#
  g0[1] = lds_byte_off;                                  // lds_addr
  g0[2] = (unsigned)(ga & 0xFFFFFFFFu);                  // global_addr lo
  g0[3] = ((unsigned)(ga >> 32) & 0x01FFFFFFu) | 0x80000000u; // hi | type=2
  i32x8 g1;
  g1[0] = (int)0x0F920000;   // data_size=4B, pad_en, interval=128dw, amount=8dw
  g1[1] = (int)0x80000000;   // tensor_dim0 = 32768 (low16 in hi half-word)
  g1[2] = (int)0x10000000;   // tensor_dim1 = 4096
  g1[3] = (int)(128 << 16);  // tile_dim0 = 128
  g1[4] = 32;                // tile_dim1 = 32
  g1[5] = 32768;             // tensor_dim0_stride = 32768 (f32 units)
  g1[6] = 0;
  g1[7] = 0;
  i32x4 z4 = {0, 0, 0, 0};               // <=2D: groups 2/3 unused
  i32x8 z8 = {0, 0, 0, 0, 0, 0, 0, 0};
  __builtin_amdgcn_tensor_load_to_lds(g0, g1, z4, z4, z8, 0);
}

// ---------------------------------------------------------------------------
// K1: row-sum of write_weights; emit xsT[f][b] = bf16(x[b][f] / sum_b) and
//     qbf[b][f] = bf16(query). One block per batch row.
// ---------------------------------------------------------------------------
__global__ __launch_bounds__(256) void k_prep(
    const float* __restrict__ W, const float* __restrict__ X,
    const float* __restrict__ Q,
    unsigned short* __restrict__ xsT, unsigned short* __restrict__ qbf) {
  __shared__ float red[256];
  const int row = blockIdx.x;
  const float* w = W + (size_t)row * MEMN;
  float s = 0.f;
  for (int j = threadIdx.x * 4; j < MEMN; j += 256 * 4) {
    float4 v = *(const float4*)(w + j);
    s += (v.x + v.y) + (v.z + v.w);
  }
  red[threadIdx.x] = s;
  __syncthreads();
  for (int off = 128; off > 0; off >>= 1) {
    if ((int)threadIdx.x < off) red[threadIdx.x] += red[threadIdx.x + off];
    __syncthreads();
  }
  const float inv = 1.0f / red[0];
  if (threadIdx.x < FDIM) {
    const int f = threadIdx.x;
    xsT[(size_t)f * BATCH + row] = f2bfu(X[(size_t)row * FDIM + f] * inv);
    qbf[(size_t)row * FDIM + f]  = f2bfu(Q[(size_t)row * FDIM + f]);
  }
}

// ---------------------------------------------------------------------------
// K2: update = W^T @ Xs ; memory_new = memory + update (f32 out + bf16 copy).
//     Block = 8 waves, 128(M) x 128(F) tile; K stepped by 32 via bf16 WMMA.
//     W slabs come in via TDM double-buffered in LDS (pitch 136).
// ---------------------------------------------------------------------------
__global__ __launch_bounds__(256) void k_update(
    const float* __restrict__ W, const float* __restrict__ Mold,
    const unsigned short* __restrict__ xsT,
    float* __restrict__ Mnew, unsigned short* __restrict__ Mbf) {
  __shared__ float ldsW[2][32 * 136];           // double-buffered slab
  const int t    = threadIdx.x;
  const int w    = t >> 5;                      // wave 0..7
  const int L    = t & 31;
  const int lmod = L & 15;
  const int m0   = blockIdx.x * 128;
  const int mloc = 16 * w + lmod;
  const int kbA  = (L < 16) ? 0 : 16;           // contiguous k order: k = 16*(L/16) + i
  const int kbB  = (L < 16) ? 0 : 16;           // contiguous k order, same as A

  v8f acc[8] = {};

  // prologue: TDM pulls slab kk=0 into buffer 0 (wave 0 only)
  if (w == 0)
    tdm_load_tile((unsigned)(uintptr_t)&ldsW[0][0], W + m0);

  int ib = 0;
  for (int kk = 0; kk < BATCH; kk += 32, ib ^= 1) {
    if (w == 0) __builtin_amdgcn_s_wait_tensorcnt(0);
    __syncthreads();                            // slab ib ready; prev reads done
    if (w == 0 && kk + 32 < BATCH)
      tdm_load_tile((unsigned)(uintptr_t)&ldsW[ib ^ 1][0],
                    W + (size_t)(kk + 32) * MEMN + m0);

    const float* buf = &ldsW[ib][0];
    // A[m,k] = W[kk+k][m0+mloc]  (transpose gather from LDS, f32 -> bf16)
    BfVec a;
    #pragma unroll
    for (int i = 0; i < 16; ++i) {
      const int k = kbA + i;
      a.s[i] = f2bfu(buf[k * 136 + mloc]);
    }
    // 8 F-tiles; B[k,f] = xsT[f][kk+k] is 32B contiguous
    #pragma unroll
    for (int nt = 0; nt < 8; ++nt) {
      const unsigned short* bp =
          xsT + (size_t)(nt * 16 + lmod) * BATCH + kk + kbB;
      acc[nt] = wmma_bf16(a.v, load_v16(bp), acc[nt]);
    }
  }

  // epilogue: C/D layout -> rows r+hi, cols = lane
  const int hi = (L >= 16) ? 8 : 0;
  #pragma unroll
  for (int nt = 0; nt < 8; ++nt) {
    #pragma unroll
    for (int r = 0; r < 8; ++r) {
      const int mrow = m0 + 16 * w + r + hi;
      const int f    = nt * 16 + lmod;
      const size_t idx = (size_t)mrow * FDIM + f;
      const float v = Mold[idx] + acc[nt][r];
      Mnew[idx] = v;
      Mbf[idx]  = f2bfu(v);
    }
  }
}

// ---------------------------------------------------------------------------
// K3: flash attention, split along the memory dimension (flash-decoding).
//     grid = (BATCH/64, SPLITS); each wave owns 16 batch rows and streams
//     MEMN/SPLITS slots in chunks of 32, emitting per-split (m, l, O).
// ---------------------------------------------------------------------------
__global__ __launch_bounds__(128) void k_attn(
    const unsigned short* __restrict__ qbf,
    const unsigned short* __restrict__ Mbf,
    float* __restrict__ Opart, float* __restrict__ mpart,
    float* __restrict__ lpart) {
  __shared__ unsigned short VT[128 * 32];       // [f][k]: transposed V chunk
  __shared__ unsigned short pA[4 * 16 * 40];    // per-wave P, pitch 40 (16B aligned)

  const int t    = threadIdx.x;
  const int w    = t >> 5;                      // wave 0..3
  const int L    = t & 31;
  const int lmod = L & 15;
  const int kbA  = (L < 16) ? 0 : 16;
  const int kbB  = (L < 16) ? 0 : 16;
  const int b0   = blockIdx.x * 64 + w * 16;
  const int sp   = blockIdx.y;
  const int mstart = sp * (MEMN / SPLITS);
  const int mend   = mstart + (MEMN / SPLITS);
  const int hi   = (L >= 16) ? 8 : 0;
  unsigned short* myP = pA + w * (16 * 40);

  // preload Q in A layout (K = 128 -> 4 chunks of 32)
  v16bf qa[4];
  {
    const unsigned short* qrow = qbf + (size_t)(b0 + lmod) * FDIM;
    #pragma unroll
    for (int kc = 0; kc < 4; ++kc)
      qa[kc] = load_a16(qrow + kc * 32 + kbA, qrow + kc * 32 + kbA + 8);
  }

  v8f O[8] = {};
  float mrow[8], lrow[8];
  #pragma unroll
  for (int r = 0; r < 8; ++r) { mrow[r] = -3.0e38f; lrow[r] = 0.0f; }

  for (int m0 = mstart; m0 < mend; m0 += 32) {
    __syncthreads();                            // protect VT rewrite
    // stage VT[f][k] = Mbf[m0+k][f]
    for (int i = 0; i < 16; ++i) {
      const int linear = i * 128 + t;           // 2048 dwords
      const int k = linear >> 6;
      const int j = linear & 63;
      const unsigned d = *(const unsigned*)(Mbf + (size_t)(m0 + k) * FDIM + 2 * j);
      VT[(2 * j) * 32 + k]     = (unsigned short)(d & 0xFFFFu);
      VT[(2 * j + 1) * 32 + k] = (unsigned short)(d >> 16);
    }
    __syncthreads();

    // scores: two 16x16 tiles over this 32-slot chunk
    v8f S0 = {}, S1 = {};
    #pragma unroll
    for (int kc = 0; kc < 4; ++kc) {
      const unsigned short* r0 =
          Mbf + (size_t)(m0 + lmod) * FDIM + kc * 32 + kbB;
      const unsigned short* r1 =
          Mbf + (size_t)(m0 + 16 + lmod) * FDIM + kc * 32 + kbB;
      S0 = wmma_bf16(qa[kc], load_v16(r0), S0);
      S1 = wmma_bf16(qa[kc], load_v16(r1), S1);
    }

    // online softmax per row (each row lives in one 16-lane half)
    #pragma unroll
    for (int r = 0; r < 8; ++r) {
      float s0 = S0[r], s1 = S1[r];
      float mx = fmaxf(s0, s1);
      #pragma unroll
      for (int d = 8; d >= 1; d >>= 1) mx = fmaxf(mx, __shfl_xor(mx, d, 16));
      const float nm    = fmaxf(mrow[r], mx);
      const float scale = __expf(mrow[r] - nm);
      const float p0 = __expf(s0 - nm);
      const float p1 = __expf(s1 - nm);
      float ps = p0 + p1;
      #pragma unroll
      for (int d = 8; d >= 1; d >>= 1) ps += __shfl_xor(ps, d, 16);
      lrow[r] = lrow[r] * scale + ps;
      mrow[r] = nm;
      #pragma unroll
      for (int ftile = 0; ftile < 8; ++ftile) O[ftile][r] *= scale;
      const int M = r + hi;
      myP[M * 40 + lmod]      = f2bfu(p0);
      myP[M * 40 + 16 + lmod] = f2bfu(p1);
    }
    __syncthreads();                            // P visible across lanes

    // O += P @ Vchunk
    const unsigned short* prow = myP + lmod * 40;
    const v16bf av = load_a16(prow + kbA, prow + kbA + 8);
    #pragma unroll
    for (int ftile = 0; ftile < 8; ++ftile) {
      const unsigned short* vp = VT + (ftile * 16 + lmod) * 32 + kbB;
      O[ftile] = wmma_bf16(av, load_v16(vp), O[ftile]);
    }
  }

  // emit per-split partials
  #pragma unroll
  for (int r = 0; r < 8; ++r) {
    const int brow = b0 + r + hi;
    #pragma unroll
    for (int ftile = 0; ftile < 8; ++ftile)
      Opart[((size_t)brow * SPLITS + sp) * FDIM + ftile * 16 + lmod] =
          O[ftile][r];
    if (lmod == 0) {
      mpart[brow * SPLITS + sp] = mrow[r];
      lpart[brow * SPLITS + sp] = lrow[r];
    }
  }
}

// ---------------------------------------------------------------------------
// K4: combine the SPLITS partials per batch row. One block per row,
//     one thread per feature column. conf = 1/l_total.
// ---------------------------------------------------------------------------
__global__ __launch_bounds__(FDIM) void k_combine(
    const float* __restrict__ Opart, const float* __restrict__ mpart,
    const float* __restrict__ lpart,
    float* __restrict__ outR, float* __restrict__ outC) {
  const int row = blockIdx.x;
  const int f   = threadIdx.x;
  float mp[SPLITS], lp[SPLITS];
  float gm = -3.0e38f;
  #pragma unroll
  for (int i = 0; i < SPLITS; ++i) {
    mp[i] = mpart[row * SPLITS + i];
    lp[i] = lpart[row * SPLITS + i];
    gm = fmaxf(gm, mp[i]);
  }
  float gl = 0.f, acc = 0.f;
  #pragma unroll
  for (int i = 0; i < SPLITS; ++i) {
    const float wgt = __expf(mp[i] - gm);
    gl  += wgt * lp[i];
    acc += wgt * Opart[((size_t)row * SPLITS + i) * FDIM + f];
  }
  const float invl = 1.0f / gl;
  outR[(size_t)row * FDIM + f] = acc * invl;
  if (f == 0) outC[row] = invl;
}

// ---------------------------------------------------------------------------
extern "C" void kernel_launch(void* const* d_in, const int* in_sizes, int n_in,
                              void* d_out, int out_size, void* d_ws, size_t ws_size,
                              hipStream_t stream) {
  (void)in_sizes; (void)n_in; (void)out_size; (void)ws_size;
  const float* memory     = (const float*)d_in[0];
  const float* input_data = (const float*)d_in[1];
  const float* write_w    = (const float*)d_in[2];
  const float* query      = (const float*)d_in[3];

  float* outR = (float*)d_out;                       // [4096,128]
  float* outC = outR + (size_t)BATCH * FDIM;         // [4096]
  float* outM = outC + BATCH;                        // [32768,128]

  unsigned short* xsT = (unsigned short*)d_ws;                 // [128][4096] bf16
  unsigned short* qbf = xsT + (size_t)FDIM * BATCH;            // [4096][128] bf16
  unsigned short* mbf = qbf + (size_t)BATCH * FDIM;            // [32768][128] bf16
  float* Opart = (float*)(mbf + (size_t)MEMN * FDIM);          // [4096][8][128] f32
  float* mpart = Opart + (size_t)BATCH * SPLITS * FDIM;        // [4096][8]
  float* lpart = mpart + (size_t)BATCH * SPLITS;               // [4096][8]

  k_prep<<<BATCH, 256, 0, stream>>>(write_w, input_data, query, xsT, qbf);
  k_update<<<MEMN / 128, 256, 0, stream>>>(write_w, memory, xsT, outM, mbf);
  dim3 g3(BATCH / 64, SPLITS);
  k_attn<<<g3, 128, 0, stream>>>(qbf, mbf, Opart, mpart, lpart);
  k_combine<<<BATCH, FDIM, 0, stream>>>(Opart, mpart, lpart, outR, outC);
}
